// RealisticDecoderLayer_20504173871419
// MI455X (gfx1250) — matrix operand order repaired and checked by reading, not compiled, not run on hardware
//
#include <hip/hip_runtime.h>
#include <hip/hip_bf16.h>

// ---------------------------------------------------------------------------
// Types for CDNA5 WMMA (wave32): bf16 A/B 16x16x32, f32 accumulator.
// ---------------------------------------------------------------------------
typedef __bf16 bf16_t;
typedef __bf16 v8bf  __attribute__((ext_vector_type(8)));
typedef __bf16 v16bf __attribute__((ext_vector_type(16)));
typedef float  v8f   __attribute__((ext_vector_type(8)));

#define MK16(lo, hi) __builtin_shufflevector((lo), (hi), 0,1,2,3,4,5,6,7,8,9,10,11,12,13,14,15)
#define AS3 __attribute__((address_space(3)))

static constexpr int   Bc  = 2;
static constexpr int   Sc  = 2048;
static constexpr int   Hc  = 2048;
static constexpr int   NHc = 16;
static constexpr int   HDc = 128;
static constexpr int   Ic  = 8192;
static constexpr int   Mc  = Bc * Sc;                     // 4096 token rows
static constexpr float EPSc = 1.1920928955078125e-07f;    // np.finfo(f32).eps

// ---------------------------------------------------------------------------
// Async global->LDS copy (16B per lane), GV mode. Per ISA §15.18.3 the
// instruction INST_OFFSET is added to BOTH the LDS address (VDST) and the
// global address (VADDR), so a compile-time byte offset rides in the
// instruction instead of costing VALU adds per chunk.
// ---------------------------------------------------------------------------
template <int OFFB>
__device__ __forceinline__ void async_copy_b128(const bf16_t* gsrc, bf16_t* ldst) {
#if defined(__gfx1250__)
    unsigned lds_off =
        (unsigned)(unsigned long long)(AS3 const bf16_t*)ldst;  // flat -> LDS offset
    asm volatile("global_load_async_to_lds_b128 %0, %1, off offset:%2"
                 :
                 : "v"(lds_off), "v"((unsigned long long)(size_t)gsrc), "n"(OFFB)
                 : "memory");
#else
    *(v8bf*)((char*)ldst + OFFB) = *(const v8bf*)((const char*)gsrc + OFFB);
#endif
}
__device__ __forceinline__ void async_wait0() {
#if defined(__gfx1250__)
#if __has_builtin(__builtin_amdgcn_s_wait_asynccnt)
    __builtin_amdgcn_s_wait_asynccnt(0);
#else
    asm volatile("s_wait_asynccnt 0" ::: "memory");
#endif
#endif
}

// ---------------------------------------------------------------------------
// fp32 -> bf16 cast (8 elements / thread, vectorized)
// ---------------------------------------------------------------------------
__global__ void __launch_bounds__(256)
cast_bf16_kernel(const float* __restrict__ in, bf16_t* __restrict__ out, size_t n) {
    size_t i      = ((size_t)blockIdx.x * blockDim.x + threadIdx.x) * 8;
    size_t stride = (size_t)gridDim.x * blockDim.x * 8;
    for (; i < n; i += stride) {
        const float4* p = (const float4*)(in + i);
        float4 a = p[0], b = p[1];
        v8bf o;
        o[0] = (bf16_t)a.x; o[1] = (bf16_t)a.y; o[2] = (bf16_t)a.z; o[3] = (bf16_t)a.w;
        o[4] = (bf16_t)b.x; o[5] = (bf16_t)b.y; o[6] = (bf16_t)b.z; o[7] = (bf16_t)b.w;
        *(v8bf*)(out + i) = o;
    }
}

// ---------------------------------------------------------------------------
// RMSNorm (fp32 in) + cast to bf16. One block (256 thr) per row of H=2048.
// ---------------------------------------------------------------------------
__global__ void __launch_bounds__(256)
rmsnorm_cast_kernel(const float* __restrict__ x, const float* __restrict__ g,
                    bf16_t* __restrict__ o, int H) {
    const int row = blockIdx.x;
    const float* xr = x + (size_t)row * H;
    float ss = 0.f;
    for (int i = threadIdx.x; i < H; i += 256) { float v = xr[i]; ss += v * v; }
#pragma unroll
    for (int m = 16; m >= 1; m >>= 1) ss += __shfl_xor(ss, m, 32);
    __shared__ float red[8];
    __shared__ float stot;
    if ((threadIdx.x & 31) == 0) red[threadIdx.x >> 5] = ss;
    __syncthreads();
    if (threadIdx.x == 0) {
        float t = 0.f;
#pragma unroll
        for (int i = 0; i < 8; ++i) t += red[i];
        stot = t;
    }
    __syncthreads();
    const float sc = rsqrtf(stot / (float)H + EPSc);
    bf16_t* orow = o + (size_t)row * H;
    for (int i = threadIdx.x; i < H; i += 256)
        orow[i] = (bf16_t)(xr[i] * sc * g[i]);
}

// ---------------------------------------------------------------------------
// silu(gate) * up, bf16 in/out, f32 math internally.
// ---------------------------------------------------------------------------
__global__ void __launch_bounds__(256)
silu_mul_kernel(const bf16_t* __restrict__ g, const bf16_t* __restrict__ u,
                bf16_t* __restrict__ h, size_t n) {
    size_t i      = ((size_t)blockIdx.x * blockDim.x + threadIdx.x) * 8;
    size_t stride = (size_t)gridDim.x * blockDim.x * 8;
    for (; i < n; i += stride) {
        v8bf gv = *(const v8bf*)(g + i);
        v8bf uv = *(const v8bf*)(u + i);
        v8bf hv;
#pragma unroll
        for (int j = 0; j < 8; ++j) {
            float x = (float)gv[j], y = (float)uv[j];
            float s = x / (1.f + __expf(-x));
            hv[j] = (bf16_t)(s * y);
        }
        *(v8bf*)(h + i) = hv;
    }
}

// ---------------------------------------------------------------------------
// GEMM: C[M,N] = A[M,K] * W[N,K]^T  (nn.Linear NT layout), bf16 in, f32 acc.
// Block = 256 threads = 8 waves; block tile 256x128, 64x64 per wave
// (wave grid 4 rows x 2 cols), 4x4 WMMA fragments.
// K-slab of 64 double-buffered through LDS via async global->LDS copies
// (ASYNCcnt): 32 v_wmma between barriers. Padded stride (72 elems) spreads
// LDS banks. RES_F32: f32 store + residual add; else bf16 store.
// ---------------------------------------------------------------------------
template <bool RES_F32>
__global__ void __launch_bounds__(256)
gemm_nt_kernel(const bf16_t* __restrict__ A, const bf16_t* __restrict__ W,
               float* __restrict__ Cf, bf16_t* __restrict__ Cb,
               const float* __restrict__ res, int M, int N, int K) {
    constexpr int LSTR = 72;                    // padded K-slab stride in elements
    __shared__ bf16_t Asm[2][256 * LSTR];       // 2 x 36 KB
    __shared__ bf16_t Bsm[2][128 * LSTR];       // 2 x 18 KB

    const int tid  = threadIdx.x;
    const int lane = tid & 31;
    const int wave = tid >> 5;
    const int nlo  = lane & 15;
    const int half = lane >> 4;

    const int brow = blockIdx.y * 256;
    const int bcol = blockIdx.x * 128;
    const int wrow = (wave & 3) * 64;
    const int wcol = (wave >> 2) * 64;

    // staging: A -> one 64-elem row per thread (8 x b128);
    //          B -> half row per thread (4 x b128)
    const int arow   = tid;
    const int brw    = tid >> 1;
    const int bchunk = (tid & 1) * 32;          // elem offset within K-slab

    const bf16_t* Ag = A + (size_t)(brow + arow) * K;
    const bf16_t* Wg = W + (size_t)(bcol + brw) * K + bchunk;

    auto stage = [&](int buf, int k0) {
        bf16_t* As = &Asm[buf][arow * LSTR];
        const bf16_t* ag = Ag + k0;
        async_copy_b128<0>(ag, As);    async_copy_b128<16>(ag, As);
        async_copy_b128<32>(ag, As);   async_copy_b128<48>(ag, As);
        async_copy_b128<64>(ag, As);   async_copy_b128<80>(ag, As);
        async_copy_b128<96>(ag, As);   async_copy_b128<112>(ag, As);
        bf16_t* Bs = &Bsm[buf][brw * LSTR + bchunk];
        const bf16_t* wg = Wg + k0;
        async_copy_b128<0>(wg, Bs);    async_copy_b128<16>(wg, Bs);
        async_copy_b128<32>(wg, Bs);   async_copy_b128<48>(wg, Bs);
    };

    v8f c[4][4] = {};

    stage(0, 0);
    async_wait0();
    __syncthreads();

    int cur = 0;
    for (int k0 = 0; k0 < K; k0 += 64) {
        if (k0 + 64 < K) stage(cur ^ 1, k0 + 64);

        const bf16_t* As = &Asm[cur][0];
        const bf16_t* Bs = &Bsm[cur][0];
#pragma unroll
        for (int ks = 0; ks < 64; ks += 32) {
            v16bf a[4], b[4];
#pragma unroll
            for (int mi = 0; mi < 4; ++mi) {
                const bf16_t* p = As + (wrow + mi * 16 + nlo) * LSTR + ks + 8 * half;
                a[mi] = MK16(*(const v8bf*)p, *(const v8bf*)(p + 16));
            }
#pragma unroll
            for (int ni = 0; ni < 4; ++ni) {
                const bf16_t* p = Bs + (wcol + ni * 16 + nlo) * LSTR + ks + 8 * half;
                b[ni] = MK16(*(const v8bf*)p, *(const v8bf*)(p + 16));
            }
#pragma unroll
            for (int mi = 0; mi < 4; ++mi)
#pragma unroll
                for (int ni = 0; ni < 4; ++ni)
                    c[mi][ni] = __builtin_amdgcn_wmma_f32_16x16x32_bf16(
                        false, a[mi], false, b[ni], (short)0, c[mi][ni], false, false);
        }

        async_wait0();      // our next-buffer fills have landed in LDS
        __syncthreads();    // everyone done reading cur / writing next
        cur ^= 1;
    }

#pragma unroll
    for (int mi = 0; mi < 4; ++mi) {
#pragma unroll
        for (int ni = 0; ni < 4; ++ni) {
#pragma unroll
            for (int r = 0; r < 8; ++r) {
                const int mrow = brow + wrow + mi * 16 + 8 * half + r;
                const int ncol = bcol + wcol + ni * 16 + nlo;
                const size_t idx = (size_t)mrow * N + ncol;
                if constexpr (RES_F32) {
                    Cf[idx] = c[mi][ni][r] + res[idx];
                } else {
                    Cb[idx] = (bf16_t)c[mi][ni][r];
                }
            }
        }
    }
}

// ---------------------------------------------------------------------------
// Flash attention: grid = (S/128, B*NH), 256 threads = 8 waves.
// Each wave owns a 16-row Q tile (Q frags resident in VGPRs).
// Per 32-row KV tile: K tile async-staged to LDS, V tile transposed into LDS;
//   S(16x32) = Q K^T (8 wmma) -> online softmax (shfl row reductions)
//   P routed through per-wave LDS scratch, re-read as A fragment
//   O(16x128) += P V (8 wmma), O rescaled by exp(m_old - m_new).
// ---------------------------------------------------------------------------
__global__ void __launch_bounds__(256)
flash_attn_kernel(const bf16_t* __restrict__ q, const bf16_t* __restrict__ k,
                  const bf16_t* __restrict__ v, bf16_t* __restrict__ o) {
    __shared__ bf16_t Kt[32 * 136];      // 32 kv rows x 128 hd (+8 pad)
    __shared__ bf16_t Vt[128 * 40];      // 128 hd rows x 32 kv (+8 pad)
    __shared__ bf16_t Pl[8 * 16 * 40];   // per-wave 16x32 P scratch (+8 pad)

    const int tid  = threadIdx.x;
    const int lane = tid & 31;
    const int wave = tid >> 5;
    const int nlo  = lane & 15;
    const int half = lane >> 4;

    const int bh   = blockIdx.y;
    const int b    = bh >> 4;            // NH = 16
    const int head = bh & 15;
    const int q0   = blockIdx.x * 128 + wave * 16;

    const bf16_t* qbase = q + ((size_t)(b * Sc + q0)) * Hc + head * HDc;
    const bf16_t* kbase = k + ((size_t)(b * Sc)) * Hc + head * HDc;
    const bf16_t* vbase = v + ((size_t)(b * Sc)) * Hc + head * HDc;

    // Preload Q as 4 A-fragments (K = 0..127).
    v16bf aq[4];
    {
        const bf16_t* p = qbase + (size_t)nlo * Hc;
#pragma unroll
        for (int kk = 0; kk < 4; ++kk) {
            v8bf lo = *(const v8bf*)(p + kk * 32 + 8 * half);
            v8bf hi = *(const v8bf*)(p + kk * 32 + 16 + 8 * half);
            aq[kk] = MK16(lo, hi);
        }
    }

    v8f   oacc[8] = {};
    float mrun[8], lrun[8];
#pragma unroll
    for (int r = 0; r < 8; ++r) { mrun[r] = -1e30f; lrun[r] = 0.f; }

    const float scale = 0.08838834764831845f;  // 1/sqrt(128)

    for (int kv0 = 0; kv0 < Sc; kv0 += 32) {
        __syncthreads();  // protect Kt/Vt WAR from previous iteration
        {   // cooperative staging: 256 thr, each does 16 elems of K and V
            const int r  = tid >> 3;          // 0..31 kv row
            const int c0 = (tid & 7) * 16;    // 0..112 hd col
            const bf16_t* kp = kbase + (size_t)(kv0 + r) * Hc + c0;
            bf16_t* kd = &Kt[r * 136 + c0];
            async_copy_b128<0>(kp, kd);
            async_copy_b128<16>(kp, kd);
            const bf16_t* vp = vbase + (size_t)(kv0 + r) * Hc + c0;
            v8bf v0 = *(const v8bf*)vp;
            v8bf v1 = *(const v8bf*)(vp + 8);
#pragma unroll
            for (int i = 0; i < 8; ++i) Vt[(c0 + i) * 40 + r]     = v0[i];
#pragma unroll
            for (int i = 0; i < 8; ++i) Vt[(c0 + 8 + i) * 40 + r] = v1[i];
        }
        async_wait0();
        __syncthreads();

        // S = Q K^T : two 16x16 f32 fragments (kv cols 0-15 and 16-31)
        v8f sf[2] = {};
#pragma unroll
        for (int ns = 0; ns < 2; ++ns) {
#pragma unroll
            for (int kk = 0; kk < 4; ++kk) {
                const bf16_t* bp = &Kt[(ns * 16 + nlo) * 136 + kk * 32 + 8 * half];
                v16bf bf = MK16(*(const v8bf*)bp, *(const v8bf*)(bp + 16));
                sf[ns] = __builtin_amdgcn_wmma_f32_16x16x32_bf16(
                    false, aq[kk], false, bf, (short)0, sf[ns], false, false);
            }
        }

        // online softmax: row max / exp / row sum (rows live in one 16-lane half)
        float alpha[8];
#pragma unroll
        for (int r = 0; r < 8; ++r) {
            float s0 = sf[0][r] * scale;
            float s1 = sf[1][r] * scale;
            float m = fmaxf(s0, s1);
            m = fmaxf(m, __shfl_xor(m, 1, 32));
            m = fmaxf(m, __shfl_xor(m, 2, 32));
            m = fmaxf(m, __shfl_xor(m, 4, 32));
            m = fmaxf(m, __shfl_xor(m, 8, 32));
            const float mnew = fmaxf(mrun[r], m);
            alpha[r] = __expf(mrun[r] - mnew);
            const float p0 = __expf(s0 - mnew);
            const float p1 = __expf(s1 - mnew);
            sf[0][r] = p0;
            sf[1][r] = p1;
            float rs = p0 + p1;
            rs += __shfl_xor(rs, 1, 32);
            rs += __shfl_xor(rs, 2, 32);
            rs += __shfl_xor(rs, 4, 32);
            rs += __shfl_xor(rs, 8, 32);
            lrun[r] = lrun[r] * alpha[r] + rs;
            mrun[r] = mnew;
        }

        // route P (C-fragment layout) -> A-fragment layout via per-wave LDS
        bf16_t* pw = &Pl[wave * 16 * 40];
#pragma unroll
        for (int r = 0; r < 8; ++r) {
            pw[(8 * half + r) * 40 + nlo]      = (bf16_t)sf[0][r];
            pw[(8 * half + r) * 40 + 16 + nlo] = (bf16_t)sf[1][r];
        }
        asm volatile("s_wait_dscnt 0" ::: "memory");  // same-wave cross-lane LDS RAW
        v16bf ap;
        {
            v8bf lo = *(const v8bf*)&pw[nlo * 40 + 8 * half];
            v8bf hi = *(const v8bf*)&pw[nlo * 40 + 16 + 8 * half];
            ap = MK16(lo, hi);
        }

        // O = O*alpha + P @ V   (V^T tile in LDS: B frag is contiguous in kv)
#pragma unroll
        for (int ni = 0; ni < 8; ++ni) {
#pragma unroll
            for (int r = 0; r < 8; ++r) oacc[ni][r] *= alpha[r];
            const bf16_t* bp = &Vt[(ni * 16 + nlo) * 40 + 8 * half];
            v16bf bv = MK16(*(const v8bf*)bp, *(const v8bf*)(bp + 16));
            oacc[ni] = __builtin_amdgcn_wmma_f32_16x16x32_bf16(
                false, ap, false, bv, (short)0, oacc[ni], false, false);
        }
    }

    // epilogue: normalize by l and store bf16 at [b, s, head*128 + d]
    bf16_t* obase = o + ((size_t)(b * Sc + q0)) * Hc + head * HDc;
#pragma unroll
    for (int ni = 0; ni < 8; ++ni) {
#pragma unroll
        for (int r = 0; r < 8; ++r) {
            const int mrow = 8 * half + r;
            obase[(size_t)mrow * Hc + ni * 16 + nlo] =
                (bf16_t)(oacc[ni][r] / lrun[r]);
        }
    }
}

// ---------------------------------------------------------------------------
// Host-side orchestration.
// Inputs: hidden_states, Wq, Wk, Wv, Wo, Wg, Wu, Wd, g_in, g_post (all f32).
// Output: f32 [B,S,H].
// ---------------------------------------------------------------------------
extern "C" void kernel_launch(void* const* d_in, const int* in_sizes, int n_in,
                              void* d_out, int out_size, void* d_ws, size_t ws_size,
                              hipStream_t stream) {
    (void)in_sizes; (void)n_in; (void)out_size; (void)ws_size;

    const float* hs     = (const float*)d_in[0];
    const float* Wq     = (const float*)d_in[1];
    const float* Wk     = (const float*)d_in[2];
    const float* Wv     = (const float*)d_in[3];
    const float* Wo     = (const float*)d_in[4];
    const float* Wg     = (const float*)d_in[5];
    const float* Wu     = (const float*)d_in[6];
    const float* Wd     = (const float*)d_in[7];
    const float* g_in   = (const float*)d_in[8];
    const float* g_post = (const float*)d_in[9];
    float* out = (float*)d_out;

    char*  ws  = (char*)d_ws;
    size_t off = 0;
    auto alloc = [&](size_t bytes) -> char* {
        char* p = ws + off;
        off += (bytes + 255) & ~(size_t)255;
        return p;
    };

    const size_t HH = (size_t)Hc * Hc, IH = (size_t)Ic * Hc;
    const size_t MH = (size_t)Mc * Hc, MI = (size_t)Mc * Ic;

    bf16_t* wq_b = (bf16_t*)alloc(HH * 2);
    bf16_t* wk_b = (bf16_t*)alloc(HH * 2);
    bf16_t* wv_b = (bf16_t*)alloc(HH * 2);
    bf16_t* wo_b = (bf16_t*)alloc(HH * 2);
    bf16_t* wg_b = (bf16_t*)alloc(IH * 2);
    bf16_t* wu_b = (bf16_t*)alloc(IH * 2);
    bf16_t* wd_b = (bf16_t*)alloc(IH * 2);
    bf16_t* xb   = (bf16_t*)alloc(MH * 2);
    bf16_t* qb   = (bf16_t*)alloc(MH * 2);
    bf16_t* kb   = (bf16_t*)alloc(MH * 2);
    bf16_t* vb   = (bf16_t*)alloc(MH * 2);
    bf16_t* attb = (bf16_t*)alloc(MH * 2);
    float*  hid2 = (float*)alloc(MH * 4);
    bf16_t* x2b  = (bf16_t*)alloc(MH * 2);
    bf16_t* gb   = (bf16_t*)alloc(MI * 2);
    bf16_t* ub   = (bf16_t*)alloc(MI * 2);
    bf16_t* hb   = (bf16_t*)alloc(MI * 2);

    auto cast = [&](const float* src, bf16_t* dst, size_t n) {
        int blocks = (int)((n / 8 + 255) / 256);
        cast_bf16_kernel<<<blocks, 256, 0, stream>>>(src, dst, n);
    };
    auto gemm_bf16 = [&](const bf16_t* A, const bf16_t* W, bf16_t* Cb,
                         int M, int N, int K) {
        dim3 grid(N / 128, M / 256);
        gemm_nt_kernel<false><<<grid, 256, 0, stream>>>(A, W, nullptr, Cb, nullptr, M, N, K);
    };
    auto gemm_f32_res = [&](const bf16_t* A, const bf16_t* W, float* Cf,
                            const float* res, int M, int N, int K) {
        dim3 grid(N / 128, M / 256);
        gemm_nt_kernel<true><<<grid, 256, 0, stream>>>(A, W, Cf, nullptr, res, M, N, K);
    };

    // weights -> bf16 (L2-resident working set: 134 MB < 192 MB L2)
    cast(Wq, wq_b, HH); cast(Wk, wk_b, HH); cast(Wv, wv_b, HH); cast(Wo, wo_b, HH);
    cast(Wg, wg_b, IH); cast(Wu, wu_b, IH); cast(Wd, wd_b, IH);

    // x = rmsnorm(hs, g_in) -> bf16
    rmsnorm_cast_kernel<<<Mc, 256, 0, stream>>>(hs, g_in, xb, Hc);

    // q/k/v projections
    gemm_bf16(xb, wq_b, qb, Mc, Hc, Hc);
    gemm_bf16(xb, wk_b, kb, Mc, Hc, Hc);
    gemm_bf16(xb, wv_b, vb, Mc, Hc, Hc);

    // attention
    flash_attn_kernel<<<dim3(Sc / 128, Bc * NHc), 256, 0, stream>>>(qb, kb, vb, attb);

    // hidden2 = hs + attn_out @ Wo^T
    gemm_f32_res(attb, wo_b, hid2, hs, Mc, Hc, Hc);

    // x2 = rmsnorm(hidden2, g_post) -> bf16
    rmsnorm_cast_kernel<<<Mc, 256, 0, stream>>>(hid2, g_post, x2b, Hc);

    // MLP
    gemm_bf16(x2b, wg_b, gb, Mc, Ic, Hc);
    gemm_bf16(x2b, wu_b, ub, Mc, Ic, Hc);
    {
        int blocks = (int)((MI / 8 + 255) / 256);
        silu_mul_kernel<<<blocks, 256, 0, stream>>>(gb, ub, hb, MI);
    }

    // out = hidden2 + h @ Wd^T
    gemm_f32_res(hb, wd_b, out, hid2, Mc, Hc, Ic);
}
